// IPAttnProcessor_33801392620175
// MI455X (gfx1250) — matrix operand order repaired and checked by reading, not compiled, not run on hardware
//
#include <hip/hip_runtime.h>
#include <stdint.h>

// ---------------------------------------------------------------------------
// CDNA5 (gfx1250) bf16 WMMA implementation of the IP-Adapter attention block.
// f32 operands are pre-converted to bf16 once; every GEMM / attention einsum
// runs through v_wmma_f32_16x16x32_bf16. GEMM tiles are staged with
// global_load_async_to_lds_b128 (ASYNCcnt) into double-buffered LDS.
// ---------------------------------------------------------------------------

typedef __attribute__((ext_vector_type(16))) __bf16 v16bf;
typedef __attribute__((ext_vector_type(2)))  __bf16 v2bf;
typedef __attribute__((ext_vector_type(8)))  float  v8f;
typedef __attribute__((ext_vector_type(2)))  float  v2f;
typedef unsigned short u16;   // bf16 storage

#define WMMA_BF16(A, B, C) \
  __builtin_amdgcn_wmma_f32_16x16x32_bf16(false, (A), false, (B), (short)0, (C), false, false)

__device__ __forceinline__ u16 f2bf(float f) {
  union { __bf16 h; u16 u; } c;
  c.h = (__bf16)f;
  return c.u;
}

__device__ __forceinline__ uint32_t pack_bf16x2(float lo, float hi) {
  v2f x; x[0] = lo; x[1] = hi;
  v2bf h = __builtin_convertvector(x, v2bf);
  union { v2bf h; uint32_t u; } c; c.h = h;
  return c.u;
}

__device__ __forceinline__ v8f vzero() {
  v8f z;
#pragma unroll
  for (int i = 0; i < 8; ++i) z[i] = 0.0f;
  return z;
}

// --- CDNA5 async global->LDS copy (16B per lane, tracked by ASYNCcnt) ------
// Immediate byte offset folded into the instruction's 24-bit IOFFSET so no
// 64-bit address VALU is needed in the inner loop.
template <int IMM>
__device__ __forceinline__ void async_copy_b128(void* lds, const void* gaddr) {
  uint32_t l = (uint32_t)(uintptr_t)lds;   // low 32 bits = LDS offset
  asm volatile("global_load_async_to_lds_b128 %0, %1, off offset:%2"
               : : "v"(l), "v"(gaddr), "n"(IMM) : "memory");
}
__device__ __forceinline__ void wait_async0() {
  asm volatile("s_wait_asynccnt 0x0" ::: "memory");
}

union FragU { uint4 q[2]; v16bf v; };

// A-matrix 16x32 bf16 fragment: lanes 0-15 -> M=lane, K=0..15;
// lanes 16-31 -> M=lane-16, K=16..31. Two b128 loads (16 consecutive K).
__device__ __forceinline__ v16bf load_frag_a(const u16* base, int stride) {
  int lane = threadIdx.x & 31;
  int row  = lane & 15;
  int klo  = (lane >> 4) << 4;            // 0 or 16
  const u16* p = base + row * stride + klo;
  FragU r;
  r.q[0] = *(const uint4*)(p);
  r.q[1] = *(const uint4*)(p + 8);
  return r.v;
}

// B-matrix 32x16 bf16 fragment, stored [N][K]: lanes 0-15 -> N=lane, K=0..15;
// lanes 16-31 -> N=lane-16, K=16..31. Two b128 loads (16 consecutive K).
__device__ __forceinline__ v16bf load_frag_b(const u16* base, int stride) {
  int lane = threadIdx.x & 31;
  int col  = lane & 15;
  int kb   = (lane >> 4) << 4;            // 0 or 16
  const u16* p = base + col * stride + kb;
  FragU r;
  r.q[0] = *(const uint4*)(p);
  r.q[1] = *(const uint4*)(p + 8);
  return r.v;
}

// Row mapping: 0 = identity, 1 = text slice of encoder states (L=81, first 77),
// 2 = ip slice (last 4).
__device__ __forceinline__ int map_row(int r, int mode) {
  if (mode == 1) { int b = r / 77; return b * 81 + (r - b * 77); }
  if (mode == 2) { int b = r >> 2; return b * 81 + 77 + (r & 3); }
  return r;
}

// ---------------------------------------------------------------------------
// Elementwise f32 -> bf16 (8 elements / thread, b128 in / b128 out)
// ---------------------------------------------------------------------------
__global__ __launch_bounds__(256)
void cvt_f32_bf16(const float* __restrict__ src, u16* __restrict__ dst, int n8)
{
  int i = blockIdx.x * 256 + threadIdx.x;
  if (i >= n8) return;
  const float4* s = (const float4*)src + (size_t)i * 2;
  float4 a = s[0], b = s[1];
  uint4 o;
  o.x = pack_bf16x2(a.x, a.y);
  o.y = pack_bf16x2(a.z, a.w);
  o.z = pack_bf16x2(b.x, b.y);
  o.w = pack_bf16x2(b.z, b.w);
  ((uint4*)dst)[i] = o;
}

// ---------------------------------------------------------------------------
// Y[M x N] (bf16) = A[M x K] (bf16, row-mapped) @ W[N x K]^T (bf16)
// 256 threads = 8 waves (2 M x 4 N), block tile 128 x 128, K-step 32.
// Double-buffered LDS, async global->LDS staging, one barrier per K-step.
// ---------------------------------------------------------------------------
__global__ __launch_bounds__(256)
void gemm_bf16(const u16* __restrict__ A, const u16* __restrict__ Bw,
               u16* __restrict__ Y, int M, int K, int N, int mode)
{
  __shared__ u16 As[2][128 * 32];
  __shared__ u16 Bs[2][128 * 32];

  const int tid  = threadIdx.x;
  const int wave = tid >> 5, lane = tid & 31;
  const int wm = wave >> 2, wn = wave & 3;
  const int m0 = blockIdx.x * 128;
  const int n0 = blockIdx.y * 128;

  v8f acc[4][2];
#pragma unroll
  for (int i = 0; i < 4; ++i) { acc[i][0] = vzero(); acc[i][1] = vzero(); }

  const int srow = tid >> 1, scol = (tid & 1) * 16;   // 16 u16 / thread / tile

  // Loop-invariant source pointers; clamp OOB rows (outputs masked at store).
  const int gr = m0 + srow;
  const int cr = (gr < M) ? gr : (M - 1);
  const u16* abase = A  + (size_t)map_row(cr, mode) * K + scol;
  const u16* bbase = Bw + (size_t)(n0 + srow) * K + scol;
  u16* aL0 = &As[0][srow * 32 + scol];
  u16* bL0 = &Bs[0][srow * 32 + scol];
  u16* aL1 = &As[1][srow * 32 + scol];
  u16* bL1 = &Bs[1][srow * 32 + scol];

  const int nK = K >> 5;

  // Preload tile 0
  async_copy_b128<0>(aL0, abase);
  async_copy_b128<16>(aL0 + 8, abase);
  async_copy_b128<0>(bL0, bbase);
  async_copy_b128<16>(bL0 + 8, bbase);
  wait_async0();
  __syncthreads();

  for (int kt = 0; kt < nK; ++kt) {
    const int cur = kt & 1;
    if (kt + 1 < nK) {                       // issue next tile (other buffer)
      const u16* ag = abase + ((kt + 1) << 5);
      const u16* bg = bbase + ((kt + 1) << 5);
      u16* aL = cur ? aL0 : aL1;
      u16* bL = cur ? bL0 : bL1;
      async_copy_b128<0>(aL, ag);
      async_copy_b128<16>(aL + 8, ag);
      async_copy_b128<0>(bL, bg);
      async_copy_b128<16>(bL + 8, bg);
    }

    v16bf b0 = load_frag_b(&Bs[cur][(wn * 32) * 32], 32);
    v16bf b1 = load_frag_b(&Bs[cur][(wn * 32 + 16) * 32], 32);
#pragma unroll
    for (int i = 0; i < 4; ++i) {
      v16bf afrag = load_frag_a(&As[cur][(wm * 64 + i * 16) * 32], 32);
      acc[i][0] = WMMA_BF16(afrag, b0, acc[i][0]);
      acc[i][1] = WMMA_BF16(afrag, b1, acc[i][1]);
    }

    if (kt + 1 < nK) wait_async0();
    __syncthreads();
  }

  const int col = lane & 15, half = lane >> 4;
#pragma unroll
  for (int i = 0; i < 4; ++i)
#pragma unroll
    for (int j = 0; j < 2; ++j)
#pragma unroll
      for (int r = 0; r < 8; ++r) {
        int m = m0 + wm * 64 + i * 16 + half * 8 + r;
        if (m < M)
          Y[(size_t)m * N + n0 + wn * 32 + j * 16 + col] = f2bf(acc[i][j][r]);
      }
}

// ---------------------------------------------------------------------------
// Y[M x N] (f32) = A[M x K] (bf16) @ W[N x K]^T (bf16) + bias + residual
// ---------------------------------------------------------------------------
__global__ __launch_bounds__(256)
void out_gemm_f32(const u16* __restrict__ A, const u16* __restrict__ Bw,
                  const float* __restrict__ bias, const float* __restrict__ res,
                  float* __restrict__ Y, int M, int K, int N)
{
  __shared__ u16 As[2][128 * 32];
  __shared__ u16 Bs[2][128 * 32];

  const int tid  = threadIdx.x;
  const int wave = tid >> 5, lane = tid & 31;
  const int wm = wave >> 2, wn = wave & 3;
  const int m0 = blockIdx.x * 128;
  const int n0 = blockIdx.y * 128;

  v8f acc[4][2];
#pragma unroll
  for (int i = 0; i < 4; ++i) { acc[i][0] = vzero(); acc[i][1] = vzero(); }

  const int srow = tid >> 1, scol = (tid & 1) * 16;
  const u16* abase = A  + (size_t)(m0 + srow) * K + scol;
  const u16* bbase = Bw + (size_t)(n0 + srow) * K + scol;
  u16* aL0 = &As[0][srow * 32 + scol];
  u16* bL0 = &Bs[0][srow * 32 + scol];
  u16* aL1 = &As[1][srow * 32 + scol];
  u16* bL1 = &Bs[1][srow * 32 + scol];

  const int nK = K >> 5;

  async_copy_b128<0>(aL0, abase);
  async_copy_b128<16>(aL0 + 8, abase);
  async_copy_b128<0>(bL0, bbase);
  async_copy_b128<16>(bL0 + 8, bbase);
  wait_async0();
  __syncthreads();

  for (int kt = 0; kt < nK; ++kt) {
    const int cur = kt & 1;
    if (kt + 1 < nK) {
      const u16* ag = abase + ((kt + 1) << 5);
      const u16* bg = bbase + ((kt + 1) << 5);
      u16* aL = cur ? aL0 : aL1;
      u16* bL = cur ? bL0 : bL1;
      async_copy_b128<0>(aL, ag);
      async_copy_b128<16>(aL + 8, ag);
      async_copy_b128<0>(bL, bg);
      async_copy_b128<16>(bL + 8, bg);
    }

    v16bf b0 = load_frag_b(&Bs[cur][(wn * 32) * 32], 32);
    v16bf b1 = load_frag_b(&Bs[cur][(wn * 32 + 16) * 32], 32);
#pragma unroll
    for (int i = 0; i < 4; ++i) {
      v16bf afrag = load_frag_a(&As[cur][(wm * 64 + i * 16) * 32], 32);
      acc[i][0] = WMMA_BF16(afrag, b0, acc[i][0]);
      acc[i][1] = WMMA_BF16(afrag, b1, acc[i][1]);
    }

    if (kt + 1 < nK) wait_async0();
    __syncthreads();
  }

  const int col = lane & 15, half = lane >> 4;
#pragma unroll
  for (int i = 0; i < 4; ++i)
#pragma unroll
    for (int j = 0; j < 2; ++j)
#pragma unroll
      for (int r = 0; r < 8; ++r) {
        int m = m0 + wm * 64 + i * 16 + half * 8 + r;
        int n = n0 + wn * 32 + j * 16 + col;
        Y[(size_t)m * N + n] = acc[i][j][r] + bias[n] + res[(size_t)m * N + n];
      }
}

// ---------------------------------------------------------------------------
// Attention: one block = 128 query rows of one (b, h).
// Text: L=77 (padded to 80), IP: L=4 (padded to 16). Softmax in registers via
// __shfl_xor over the 16-lane halves; probs staged to LDS for the P@V WMMAs.
// ---------------------------------------------------------------------------
__global__ __launch_bounds__(256)
void attn_kernel(const u16* __restrict__ Q,  const u16* __restrict__ Kt,
                 const u16* __restrict__ Vt, const u16* __restrict__ Kip,
                 const u16* __restrict__ Vip, u16* __restrict__ O,
                 float scale, float ip_scale)
{
  __shared__ u16 sK[80 * 64];        // K text  [l][d]
  __shared__ u16 sVt[64 * 96];       // V text transposed [d][l], l padded to 96
  __shared__ u16 sKip[16 * 64];      // K ip    [l][d]
  __shared__ u16 sVipT[64 * 32];     // V ip transposed [d][l], l padded to 32
  __shared__ u16 sP[8][16 * 96];     // per-wave probs (text)
  __shared__ u16 sPip[8][16 * 32];   // per-wave probs (ip)

  const int tid = threadIdx.x, wave = tid >> 5, lane = tid & 31;
  const int bh = blockIdx.y, b = bh / 20, h = bh % 20;
  const int C = 1280, S = 4096;
  const int qbase = blockIdx.x * 128;

  for (int idx = tid; idx < 80 * 64; idx += 256) {
    int l = idx >> 6, d = idx & 63;
    sK[idx] = (l < 77) ? Kt[(size_t)(b * 77 + l) * C + h * 64 + d] : (u16)0;
  }
  for (int idx = tid; idx < 64 * 96; idx += 256) {
    int d = idx / 96, l = idx % 96;
    sVt[idx] = (l < 77) ? Vt[(size_t)(b * 77 + l) * C + h * 64 + d] : (u16)0;
  }
  for (int idx = tid; idx < 16 * 64; idx += 256) {
    int l = idx >> 6, d = idx & 63;
    sKip[idx] = (l < 4) ? Kip[(size_t)(b * 4 + l) * C + h * 64 + d] : (u16)0;
  }
  for (int idx = tid; idx < 64 * 32; idx += 256) {
    int d = idx >> 5, l = idx & 31;
    sVipT[idx] = (l < 4) ? Vip[(size_t)(b * 4 + l) * C + h * 64 + d] : (u16)0;
  }
  __syncthreads();

  // Q tile (16x64) as two A fragments (K-steps of 32), straight from global.
  const u16* qptr = Q + (size_t)(b * S + qbase + wave * 16) * C + h * 64;
  v16bf aq0 = load_frag_a(qptr, C);
  v16bf aq1 = load_frag_a(qptr + 32, C);

  // scores: S = Q K^T
  v8f st[5];
#pragma unroll
  for (int t = 0; t < 5; ++t) {
    v8f s = vzero();
    s = WMMA_BF16(aq0, load_frag_b(&sK[(t * 16) * 64], 64), s);
    s = WMMA_BF16(aq1, load_frag_b(&sK[(t * 16) * 64 + 32], 64), s);
    st[t] = s;
  }
  v8f sip = vzero();
  sip = WMMA_BF16(aq0, load_frag_b(&sKip[0], 64), sip);
  sip = WMMA_BF16(aq1, load_frag_b(&sKip[32], 64), sip);

  const int col = lane & 15, half = lane >> 4;
  const float NEG = -3.0e38f;

  float rmax[8], rsum[8], imax[8], isum[8];
#pragma unroll
  for (int r = 0; r < 8; ++r) { rmax[r] = NEG; rsum[r] = 0.0f; isum[r] = 0.0f; }

#pragma unroll
  for (int t = 0; t < 5; ++t) {
    int n = t * 16 + col;
#pragma unroll
    for (int r = 0; r < 8; ++r) {
      float x = st[t][r] * scale;
      if (n >= 77) x = NEG;
      st[t][r] = x;
      rmax[r] = fmaxf(rmax[r], x);
    }
  }
#pragma unroll
  for (int r = 0; r < 8; ++r) {
    float x = sip[r] * scale;
    if (col >= 4) x = NEG;
    sip[r] = x;
    imax[r] = x;
  }
#pragma unroll
  for (int off = 1; off < 16; off <<= 1) {
#pragma unroll
    for (int r = 0; r < 8; ++r) {
      rmax[r] = fmaxf(rmax[r], __shfl_xor(rmax[r], off, 32));
      imax[r] = fmaxf(imax[r], __shfl_xor(imax[r], off, 32));
    }
  }
#pragma unroll
  for (int t = 0; t < 5; ++t)
#pragma unroll
    for (int r = 0; r < 8; ++r) {
      float p = __expf(st[t][r] - rmax[r]);
      st[t][r] = p;
      rsum[r] += p;
    }
#pragma unroll
  for (int r = 0; r < 8; ++r) {
    float p = __expf(sip[r] - imax[r]);
    sip[r] = p;
    isum[r] += p;
  }
#pragma unroll
  for (int off = 1; off < 16; off <<= 1) {
#pragma unroll
    for (int r = 0; r < 8; ++r) {
      rsum[r] += __shfl_xor(rsum[r], off, 32);
      isum[r] += __shfl_xor(isum[r], off, 32);
    }
  }

  // stage unnormalized probs (bf16) as A-matrices for P@V
  u16* pw  = &sP[wave][0];
  u16* pip = &sPip[wave][0];
#pragma unroll
  for (int t = 0; t < 5; ++t)
#pragma unroll
    for (int r = 0; r < 8; ++r)
      pw[(half * 8 + r) * 96 + t * 16 + col] = f2bf(st[t][r]);
#pragma unroll
  for (int j = 0; j < 8; ++j) {                 // zero pad cols 80..95
    int idx = lane * 8 + j;
    pw[(idx >> 4) * 96 + 80 + (idx & 15)] = 0;
  }
#pragma unroll
  for (int r = 0; r < 8; ++r)
    pip[(half * 8 + r) * 32 + col] = f2bf(sip[r]);
#pragma unroll
  for (int j = 0; j < 8; ++j) {                 // zero pad cols 16..31
    int idx = lane * 8 + j;
    pip[(idx >> 4) * 32 + 16 + (idx & 15)] = 0;
  }

  // O = P @ V   (text: K=96 -> 3 steps; ip: K=32 -> 1 step), 4 N-tiles of 16
  v8f ot[4], oi[4];
#pragma unroll
  for (int nt = 0; nt < 4; ++nt) {
    v8f o = vzero();
#pragma unroll
    for (int ks = 0; ks < 3; ++ks)
      o = WMMA_BF16(load_frag_a(pw + ks * 32, 96),
                    load_frag_b(&sVt[(nt * 16) * 96 + ks * 32], 96), o);
    ot[nt] = o;
    v8f o2 = vzero();
    o2 = WMMA_BF16(load_frag_a(pip, 32),
                   load_frag_b(&sVipT[(nt * 16) * 32], 32), o2);
    oi[nt] = o2;
  }

  float rinv[8], iinv[8];
#pragma unroll
  for (int r = 0; r < 8; ++r) { rinv[r] = 1.0f / rsum[r]; iinv[r] = 1.0f / isum[r]; }

  const size_t orow0 = (size_t)(b * S + qbase + wave * 16);
#pragma unroll
  for (int nt = 0; nt < 4; ++nt)
#pragma unroll
    for (int r = 0; r < 8; ++r) {
      int m = half * 8 + r;
      float v = ot[nt][r] * rinv[r] + ip_scale * oi[nt][r] * iinv[r];
      O[(orow0 + m) * C + h * 64 + nt * 16 + col] = f2bf(v);
    }
}

// ---------------------------------------------------------------------------
extern "C" void kernel_launch(void* const* d_in, const int* in_sizes, int n_in,
                              void* d_out, int out_size, void* d_ws, size_t ws_size,
                              hipStream_t stream) {
  (void)in_sizes; (void)n_in; (void)out_size; (void)ws_size;
  const float* hs    = (const float*)d_in[0];   // (8, 4096, 1280)
  const float* ehs   = (const float*)d_in[1];   // (8, 81, 2048)
  const float* Wq    = (const float*)d_in[2];
  const float* Wk    = (const float*)d_in[3];
  const float* Wv    = (const float*)d_in[4];
  const float* Wk_ip = (const float*)d_in[5];
  const float* Wv_ip = (const float*)d_in[6];
  const float* Wo    = (const float*)d_in[7];
  const float* bo    = (const float*)d_in[8];
  float* out = (float*)d_out;

  const int B = 8, S = 4096, C = 1280, CC = 2048, L = 81;
  const int M  = B * S;          // 32768
  const int Mt = B * 77;         // 616
  const int Mi = B * 4;          // 32

  char* ws = (char*)d_ws;
  size_t off = 0;
  auto alloc = [&](size_t bytes) -> void* {
    void* p = ws + off;
    off = (off + bytes + 255) & ~(size_t)255;
    return p;
  };
  u16* hs_bf  = (u16*)alloc((size_t)M * C * 2);
  u16* ehs_bf = (u16*)alloc((size_t)B * L * CC * 2);
  u16* wq_bf  = (u16*)alloc((size_t)C * C * 2);
  u16* wk_bf  = (u16*)alloc((size_t)C * CC * 2);
  u16* wv_bf  = (u16*)alloc((size_t)C * CC * 2);
  u16* wki_bf = (u16*)alloc((size_t)C * CC * 2);
  u16* wvi_bf = (u16*)alloc((size_t)C * CC * 2);
  u16* wo_bf  = (u16*)alloc((size_t)C * C * 2);
  u16* q_bf   = (u16*)alloc((size_t)M  * C * 2);
  u16* o_bf   = (u16*)alloc((size_t)M  * C * 2);
  u16* k_text = (u16*)alloc((size_t)Mt * C * 2);
  u16* v_text = (u16*)alloc((size_t)Mt * C * 2);
  u16* k_ip   = (u16*)alloc((size_t)Mi * C * 2);
  u16* v_ip   = (u16*)alloc((size_t)Mi * C * 2);

  dim3 blk(256);
  auto cvt = [&](const float* s, u16* d, size_t n) {
    int n8 = (int)(n / 8);
    cvt_f32_bf16<<<dim3((n8 + 255) / 256), blk, 0, stream>>>(s, d, n8);
  };
  cvt(hs,    hs_bf,  (size_t)M * C);
  cvt(ehs,   ehs_bf, (size_t)B * L * CC);
  cvt(Wq,    wq_bf,  (size_t)C * C);
  cvt(Wk,    wk_bf,  (size_t)C * CC);
  cvt(Wv,    wv_bf,  (size_t)C * CC);
  cvt(Wk_ip, wki_bf, (size_t)C * CC);
  cvt(Wv_ip, wvi_bf, (size_t)C * CC);
  cvt(Wo,    wo_bf,  (size_t)C * C);

  // projections (bf16 x bf16 -> bf16)
  gemm_bf16<<<dim3(M / 128, C / 128), blk, 0, stream>>>(hs_bf,  wq_bf,  q_bf,   M,  C,  C, 0);
  gemm_bf16<<<dim3((Mt + 127) / 128, C / 128), blk, 0, stream>>>(ehs_bf, wk_bf,  k_text, Mt, CC, C, 1);
  gemm_bf16<<<dim3((Mt + 127) / 128, C / 128), blk, 0, stream>>>(ehs_bf, wv_bf,  v_text, Mt, CC, C, 1);
  gemm_bf16<<<dim3(1, C / 128), blk, 0, stream>>>(ehs_bf, wki_bf, k_ip,   Mi, CC, C, 2);
  gemm_bf16<<<dim3(1, C / 128), blk, 0, stream>>>(ehs_bf, wvi_bf, v_ip,   Mi, CC, C, 2);

  // attention (text + ip), merged heads, bf16 out
  attn_kernel<<<dim3(S / 128, B * 20), blk, 0, stream>>>(
      q_bf, k_text, v_text, k_ip, v_ip, o_bf, 0.125f /* 1/sqrt(64) */, 1.0f);

  // output projection + bias + residual (f32)
  out_gemm_f32<<<dim3(M / 128, C / 128), blk, 0, stream>>>(o_bf, wo_bf, bo, hs, out, M, C, C);
}
